// CausalSelfAttention_29669634081241
// MI455X (gfx1250) — matrix operand order repaired and checked by reading, not compiled, not run on hardware
//
#include <hip/hip_runtime.h>
#include <stdint.h>

// ---------------------------------------------------------------------------
// Causal self-attention for MI455X (gfx1250, wave32, WMMA).
// B=2, L=4096, D=768, H=12, Dh=64.
// Pipeline: QKV GEMM (bf16 WMMA) -> flash attention (bf16 WMMA, fp32 softmax)
//           -> output GEMM (bf16 WMMA, fp32 out).
// bf16-only staging streams use GLOBAL_LOAD_ASYNC_TO_LDS_B128 (ASYNCcnt).
// ---------------------------------------------------------------------------

typedef __bf16 bf16;
typedef __attribute__((ext_vector_type(16))) __bf16 v16bf;
typedef __attribute__((ext_vector_type(8)))  float  v8f;

#define B_   2
#define L_   4096
#define D_   768
#define H_   12
#define DH_  64

union FragBF { v16bf v; uint4 q[2]; };
union Acc8   { v8f v; float f[8]; };

__device__ __forceinline__ v8f wmma_bf16(const FragBF& a, const FragBF& b, v8f c) {
  // D = A(16x32 bf16) * B(32x16 bf16) + C(16x16 f32)
  return __builtin_amdgcn_wmma_f32_16x16x32_bf16(false, a.v, false, b.v,
                                                 (short)0, c, false, false);
}

// Async copy 16B global -> LDS.  ISA: IOFFSET is added to BOTH the LDS
// destination and the global source address, so one (ldsaddr, voffset) pair
// can issue several contiguous chunks with different immediates.
#define ASYNC_B128(ldsaddr, voff, sbase, imm)                                  \
  asm volatile("global_load_async_to_lds_b128 %0, %1, %2 offset:" #imm         \
               :: "v"(ldsaddr), "v"(voff), "s"(sbase) : "memory")

__device__ __forceinline__ void wait_async0() {
  asm volatile("s_wait_asynccnt 0x0" ::: "memory");
}

__device__ __forceinline__ uint32_t lds_lo32(const void* p) {
  // Generic pointers to LDS carry the LDS byte offset in the low 32 bits
  // (flat-address aperture rule).
  return (uint32_t)(uintptr_t)p;
}

// ---------------------------------------------------------------------------
// Kernel 1: fused QKV projection.  C[8192 x 2304] = x[8192 x 768] @ Wqkv + b.
// Block tile 128x128, K-step 32.  8 waves, each wave owns a 32x64 sub-tile
// (2 M-subtiles x 4 N-subtiles of 16x16).  Staging converts fp32 -> bf16 in
// flight (VALU path).  Epilogue scatters into per-head bf16 Q (pre-scaled by
// 1/sqrt(Dh)=0.125), K [B,H,L,Dh], and V^T [B,H,Dh,L].
// ---------------------------------------------------------------------------
__global__ __launch_bounds__(256)
void qkv_proj_wmma(const float* __restrict__ x, const float* __restrict__ W,
                   const float* __restrict__ bias,
                   bf16* __restrict__ qws, bf16* __restrict__ kws,
                   bf16* __restrict__ vws)
{
  __shared__ bf16 As[128 * 32];   // [m][k] row-major
  __shared__ bf16 Bs[128 * 32];   // [n][k] (transposed weight tile)

  const int tid  = threadIdx.x;
  const int lane = tid & 31;
  const int l15  = lane & 15;
  const int lhi  = lane >> 4;
  const int wv   = tid >> 5;
  const int wm   = wv & 3;        // 4 wave rows
  const int wn   = wv >> 2;       // 2 wave cols
  const int m0   = blockIdx.y * 128;
  const int n0   = blockIdx.x * 128;

  v8f acc[2][4];
#pragma unroll
  for (int i = 0; i < 2; ++i)
#pragma unroll
    for (int j = 0; j < 4; ++j)
#pragma unroll
      for (int e = 0; e < 8; ++e) acc[i][j][e] = 0.0f;

  const int ar = tid >> 1;            // A stage: row 0..127
  const int ac = (tid & 1) * 16;      // A stage: k col 0 or 16
  const int bk = tid >> 3;            // B stage: k row 0..31
  const int bn = (tid & 7) * 16;      // B stage: n col base

  for (int k0 = 0; k0 < D_; k0 += 32) {
    // ---- stage A: fp32 -> bf16, 128x32
    {
      const float4* src = (const float4*)(x + (size_t)(m0 + ar) * D_ + k0 + ac);
      float4 f0 = src[0], f1 = src[1], f2 = src[2], f3 = src[3];
      union { bf16 h[16]; uint4 q[2]; } pk;
      pk.h[0]=(bf16)f0.x; pk.h[1]=(bf16)f0.y; pk.h[2]=(bf16)f0.z; pk.h[3]=(bf16)f0.w;
      pk.h[4]=(bf16)f1.x; pk.h[5]=(bf16)f1.y; pk.h[6]=(bf16)f1.z; pk.h[7]=(bf16)f1.w;
      pk.h[8]=(bf16)f2.x; pk.h[9]=(bf16)f2.y; pk.h[10]=(bf16)f2.z; pk.h[11]=(bf16)f2.w;
      pk.h[12]=(bf16)f3.x; pk.h[13]=(bf16)f3.y; pk.h[14]=(bf16)f3.z; pk.h[15]=(bf16)f3.w;
      uint4* dst = (uint4*)&As[ar * 32 + ac];
      dst[0] = pk.q[0]; dst[1] = pk.q[1];
    }
    // ---- stage B transposed: Bs[n][k] = W[k0+k][n0+n], fp32 -> bf16
    {
      const float4* src = (const float4*)(W + (size_t)(k0 + bk) * (3 * D_) + n0 + bn);
      float4 f0 = src[0], f1 = src[1], f2 = src[2], f3 = src[3];
      bf16 hv[16] = {(bf16)f0.x,(bf16)f0.y,(bf16)f0.z,(bf16)f0.w,
                     (bf16)f1.x,(bf16)f1.y,(bf16)f1.z,(bf16)f1.w,
                     (bf16)f2.x,(bf16)f2.y,(bf16)f2.z,(bf16)f2.w,
                     (bf16)f3.x,(bf16)f3.y,(bf16)f3.z,(bf16)f3.w};
#pragma unroll
      for (int i = 0; i < 16; ++i) Bs[(bn + i) * 32 + bk] = hv[i];
    }
    __syncthreads();

    // A fragment layout (16-bit A 16x32): lanes 0-15 -> K {0..7,16..23},
    // lanes 16-31 -> K {8..15,24..31}
    FragBF af[2];
#pragma unroll
    for (int ms = 0; ms < 2; ++ms) {
      const bf16* ap = &As[(wm * 32 + ms * 16 + l15) * 32];
      af[ms].q[0] = *(const uint4*)(ap + lhi * 8);
      af[ms].q[1] = *(const uint4*)(ap + 16 + lhi * 8);
    }
#pragma unroll
    for (int ns = 0; ns < 4; ++ns) {
      // B fragment: lane = column N, K {0..7,16..23} / {8..15,24..31} (same as A)
      FragBF bfr;
      const bf16* bp = &Bs[(wn * 64 + ns * 16 + l15) * 32];
      bfr.q[0] = *(const uint4*)(bp + lhi * 8);
      bfr.q[1] = *(const uint4*)(bp + 16 + lhi * 8);
#pragma unroll
      for (int ms = 0; ms < 2; ++ms)
        acc[ms][ns] = wmma_bf16(af[ms], bfr, acc[ms][ns]);
    }
    __syncthreads();
  }

  // ---- epilogue: bias + scatter (C layout: VGPR r = row, lane&15 = col)
#pragma unroll
  for (int ms = 0; ms < 2; ++ms)
#pragma unroll
    for (int ns = 0; ns < 4; ++ns) {
      Acc8 a; a.v = acc[ms][ns];
      const int col = n0 + wn * 64 + ns * 16 + l15;
      const float bj = bias[col];
#pragma unroll
      for (int r = 0; r < 8; ++r) {
        const int row = m0 + wm * 32 + ms * 16 + lhi * 8 + r;
        const float val = a.f[r] + bj;
        const int b = row >> 12;
        const int l = row & (L_ - 1);
        if (col < D_) {
          const int h = col >> 6, dh = col & 63;
          qws[((size_t)((b * H_ + h) * L_ + l) << 6) + dh] = (bf16)(val * 0.125f);
        } else if (col < 2 * D_) {
          const int c = col - D_;
          const int h = c >> 6, dh = c & 63;
          kws[((size_t)((b * H_ + h) * L_ + l) << 6) + dh] = (bf16)val;
        } else {
          const int c = col - 2 * D_;
          const int h = c >> 6, dh = c & 63;
          vws[(size_t)((b * H_ + h) * DH_ + dh) * L_ + l] = (bf16)val;
        }
      }
    }
}

// ---------------------------------------------------------------------------
// Kernel 2: causal flash attention.  One block = (b,h) x 64 query rows,
// 4 waves x 16 rows each.  K/V^T tiles (pure bf16 bytes) staged to LDS with
// GLOBAL_LOAD_ASYNC_TO_LDS_B128 + s_wait_asynccnt; online softmax in fp32
// with shfl_xor row reductions; P re-swizzled to A-fragment layout through a
// per-wave LDS tile.  8 WMMAs per wave per key tile.
// ---------------------------------------------------------------------------
__global__ __launch_bounds__(128)
void flash_attn_wmma(const bf16* __restrict__ qws, const bf16* __restrict__ kws,
                     const bf16* __restrict__ vws, bf16* __restrict__ yws)
{
  __shared__ bf16 Kt[32 * 64];      // [key][dh]
  __shared__ bf16 Vt[64 * 32];      // [dh][key]  (V already transposed in ws)
  __shared__ bf16 Pl[4 * 16 * 32];  // per-wave P tile [16 rows][32 keys]

  const int tid = threadIdx.x;
  const int lane = tid & 31;
  const int l15 = lane & 15;
  const int lhi = lane >> 4;
  const int w   = tid >> 5;
  const int q0  = blockIdx.x * 64;
  const int bh  = blockIdx.y;
  const int b   = bh / H_;
  const int h   = bh % H_;

  const bf16* Qb = qws + (size_t)bh * L_ * DH_;
  const bf16* Kb = kws + (size_t)bh * L_ * DH_;
  const bf16* Vb = vws + (size_t)bh * DH_ * L_;
  const uint64_t Kb64 = (uint64_t)(uintptr_t)Kb;
  const uint64_t Vb64 = (uint64_t)(uintptr_t)Vb;

  // Per-thread LDS dest addresses / global voffsets for async staging.
  const uint32_t ldsK = lds_lo32(Kt) + (uint32_t)tid * 16;
  const int vdh = tid >> 1, vpart = tid & 1;
  const uint32_t ldsV = lds_lo32(Vt) + (uint32_t)(vdh * 64 + vpart * 32);

  // Q fragments (held in registers for the whole KV loop); Q pre-scaled.
  FragBF qa[2];
  {
    const bf16* qp = Qb + (size_t)(q0 + w * 16 + l15) * DH_;
    qa[0].q[0] = *(const uint4*)(qp + lhi * 8);
    qa[0].q[1] = *(const uint4*)(qp + 16 + lhi * 8);
    qa[1].q[0] = *(const uint4*)(qp + 32 + lhi * 8);
    qa[1].q[1] = *(const uint4*)(qp + 48 + lhi * 8);
  }

  float mrow[8], lrow[8];
  v8f O[4];
#pragma unroll
  for (int r = 0; r < 8; ++r) { mrow[r] = -1e30f; lrow[r] = 0.0f; }
#pragma unroll
  for (int ns = 0; ns < 4; ++ns)
#pragma unroll
    for (int e = 0; e < 8; ++e) O[ns][e] = 0.0f;

  const int nkt = q0 / 32 + 2;   // causal: keys up to q0+63
  for (int kt = 0; kt < nkt; ++kt) {
    const int k0 = kt * 32;
    // ---- async stage K tile: 32 keys x 64 dh = 4KB contiguous in global
    {
      const uint32_t goffK = (uint32_t)(k0 * (DH_ * 2)) + (uint32_t)tid * 16;
      ASYNC_B128(ldsK, goffK, Kb64, 0);
      ASYNC_B128(ldsK, goffK, Kb64, 2048);
    }
    // ---- async stage V^T tile: 64 dh rows x 32 keys (64B per row)
    {
      const uint32_t goffV =
          (uint32_t)(vdh * (L_ * 2) + k0 * 2 + vpart * 32);
      ASYNC_B128(ldsV, goffV, Vb64, 0);
      ASYNC_B128(ldsV, goffV, Vb64, 16);
    }
    // warm L2 with the next K tile while WMMAs run
    if (kt + 1 < nkt)
      __builtin_prefetch(Kb + (size_t)(k0 + 32) * DH_ + tid * 8, 0, 1);
    wait_async0();
    __syncthreads();

    // S = Q K^T : 16 rows x 32 keys (two 16-key subtiles, Dh split 32+32)
    v8f S[2];
#pragma unroll
    for (int sn = 0; sn < 2; ++sn) {
#pragma unroll
      for (int e = 0; e < 8; ++e) S[sn][e] = 0.0f;
      FragBF kb0, kb1;
      const bf16* kp = Kt + (sn * 16 + l15) * 64;
      kb0.q[0] = *(const uint4*)(kp + lhi * 8);
      kb0.q[1] = *(const uint4*)(kp + 16 + lhi * 8);
      kb1.q[0] = *(const uint4*)(kp + 32 + lhi * 8);
      kb1.q[1] = *(const uint4*)(kp + 48 + lhi * 8);
      S[sn] = wmma_bf16(qa[0], kb0, S[sn]);
      S[sn] = wmma_bf16(qa[1], kb1, S[sn]);
    }

    // online softmax (fp32): C layout -> lane&15 = key col, VGPR r = row
    Acc8 s0, s1; s0.v = S[0]; s1.v = S[1];
    const int rowbase = q0 + w * 16 + lhi * 8;
    const int key0 = k0 + l15;
    const int key1 = k0 + 16 + l15;
#pragma unroll
    for (int r = 0; r < 8; ++r) {
      const int row = rowbase + r;
      float v0 = (key0 <= row) ? s0.f[r] : -1e30f;
      float v1 = (key1 <= row) ? s1.f[r] : -1e30f;
      float t = fmaxf(v0, v1);
      t = fmaxf(t, __shfl_xor(t, 1));
      t = fmaxf(t, __shfl_xor(t, 2));
      t = fmaxf(t, __shfl_xor(t, 4));
      t = fmaxf(t, __shfl_xor(t, 8));
      const float mnew  = fmaxf(mrow[r], t);
      const float scale = __expf(mrow[r] - mnew);
      mrow[r] = mnew;
      const float p0 = __expf(v0 - mnew);
      const float p1 = __expf(v1 - mnew);
      float ps = p0 + p1;
      ps += __shfl_xor(ps, 1);
      ps += __shfl_xor(ps, 2);
      ps += __shfl_xor(ps, 4);
      ps += __shfl_xor(ps, 8);
      lrow[r] = lrow[r] * scale + ps;
      O[0][r] *= scale; O[1][r] *= scale; O[2][r] *= scale; O[3][r] *= scale;
      s0.f[r] = p0; s1.f[r] = p1;
    }

    // P -> bf16 -> per-wave LDS tile -> A fragment; then PV WMMAs
    {
      bf16* pw = Pl + w * 16 * 32;
#pragma unroll
      for (int r = 0; r < 8; ++r) {
        pw[(lhi * 8 + r) * 32 + l15]      = (bf16)s0.f[r];
        pw[(lhi * 8 + r) * 32 + 16 + l15] = (bf16)s1.f[r];
      }
      FragBF pf;
      const bf16* pp = pw + l15 * 32;
      pf.q[0] = *(const uint4*)(pp + lhi * 8);
      pf.q[1] = *(const uint4*)(pp + 16 + lhi * 8);
#pragma unroll
      for (int ns = 0; ns < 4; ++ns) {
        FragBF vf;
        const bf16* vp = Vt + (ns * 16 + l15) * 32;
        vf.q[0] = *(const uint4*)(vp + lhi * 8);
        vf.q[1] = *(const uint4*)(vp + 16 + lhi * 8);
        O[ns] = wmma_bf16(pf, vf, O[ns]);
      }
    }
    __syncthreads();
  }

  // normalize rows and emit Y (bf16) in [B][L][D] layout for the final GEMM
  float inv[8];
#pragma unroll
  for (int r = 0; r < 8; ++r) inv[r] = 1.0f / lrow[r];
#pragma unroll
  for (int ns = 0; ns < 4; ++ns)
#pragma unroll
    for (int r = 0; r < 8; ++r) {
      const int row = q0 + w * 16 + lhi * 8 + r;
      const int col = h * DH_ + ns * 16 + l15;
      yws[(size_t)(b * L_ + row) * D_ + col] = (bf16)(O[ns][r] * inv[r]);
    }
}

// ---------------------------------------------------------------------------
// Kernel 3: output projection.  out[8192 x 768] = Y(bf16) @ Wo + bo, fp32 out.
// A tile (already bf16) staged with async-to-LDS; B tile converts fp32->bf16.
// ---------------------------------------------------------------------------
__global__ __launch_bounds__(256)
void out_proj_wmma(const bf16* __restrict__ Y, const float* __restrict__ Wo,
                   const float* __restrict__ bo, float* __restrict__ out)
{
  __shared__ bf16 As[128 * 32];
  __shared__ bf16 Bs[128 * 32];

  const int tid  = threadIdx.x;
  const int lane = tid & 31;
  const int l15  = lane & 15;
  const int lhi  = lane >> 4;
  const int wv   = tid >> 5;
  const int wm   = wv & 3;
  const int wn   = wv >> 2;
  const int m0   = blockIdx.y * 128;
  const int n0   = blockIdx.x * 128;

  v8f acc[2][4];
#pragma unroll
  for (int i = 0; i < 2; ++i)
#pragma unroll
    for (int j = 0; j < 4; ++j)
#pragma unroll
      for (int e = 0; e < 8; ++e) acc[i][j][e] = 0.0f;

  const int ar = tid >> 1;
  const int ac = (tid & 1) * 16;
  const int bk = tid >> 3;
  const int bn = (tid & 7) * 16;

  const uint64_t Y64 = (uint64_t)(uintptr_t)Y;
  const uint32_t ldsA = lds_lo32(As) + (uint32_t)((ar * 32 + ac) * 2);

  for (int k0 = 0; k0 < D_; k0 += 32) {
    // ---- async stage A (bf16 bytes straight from Y): 32B per thread
    {
      const uint32_t goffA = (uint32_t)(((m0 + ar) * D_ + k0 + ac) * 2);
      ASYNC_B128(ldsA, goffA, Y64, 0);
      ASYNC_B128(ldsA, goffA, Y64, 16);
    }
    // ---- stage B transposed: Bs[n][k] = Wo[k0+k][n0+n], fp32 -> bf16
    {
      const float4* src = (const float4*)(Wo + (size_t)(k0 + bk) * D_ + n0 + bn);
      float4 f0 = src[0], f1 = src[1], f2 = src[2], f3 = src[3];
      bf16 hv[16] = {(bf16)f0.x,(bf16)f0.y,(bf16)f0.z,(bf16)f0.w,
                     (bf16)f1.x,(bf16)f1.y,(bf16)f1.z,(bf16)f1.w,
                     (bf16)f2.x,(bf16)f2.y,(bf16)f2.z,(bf16)f2.w,
                     (bf16)f3.x,(bf16)f3.y,(bf16)f3.z,(bf16)f3.w};
#pragma unroll
      for (int i = 0; i < 16; ++i) Bs[(bn + i) * 32 + bk] = hv[i];
    }
    wait_async0();
    __syncthreads();

    FragBF af[2];
#pragma unroll
    for (int ms = 0; ms < 2; ++ms) {
      const bf16* ap = &As[(wm * 32 + ms * 16 + l15) * 32];
      af[ms].q[0] = *(const uint4*)(ap + lhi * 8);
      af[ms].q[1] = *(const uint4*)(ap + 16 + lhi * 8);
    }
#pragma unroll
    for (int ns = 0; ns < 4; ++ns) {
      FragBF bfr;
      const bf16* bp = &Bs[(wn * 64 + ns * 16 + l15) * 32];
      bfr.q[0] = *(const uint4*)(bp + lhi * 8);
      bfr.q[1] = *(const uint4*)(bp + 16 + lhi * 8);
#pragma unroll
      for (int ms = 0; ms < 2; ++ms)
        acc[ms][ns] = wmma_bf16(af[ms], bfr, acc[ms][ns]);
    }
    __syncthreads();
  }

#pragma unroll
  for (int ms = 0; ms < 2; ++ms)
#pragma unroll
    for (int ns = 0; ns < 4; ++ns) {
      Acc8 a; a.v = acc[ms][ns];
      const int col = n0 + wn * 64 + ns * 16 + l15;
      const float bj = bo[col];
#pragma unroll
      for (int r = 0; r < 8; ++r) {
        const int row = m0 + wm * 32 + ms * 16 + lhi * 8 + r;
        out[(size_t)row * D_ + col] = a.f[r] + bj;
      }
    }
}

// ---------------------------------------------------------------------------
extern "C" void kernel_launch(void* const* d_in, const int* in_sizes, int n_in,
                              void* d_out, int out_size, void* d_ws, size_t ws_size,
                              hipStream_t stream) {
  const float* x    = (const float*)d_in[0];
  const float* Wqkv = (const float*)d_in[1];
  const float* bqkv = (const float*)d_in[2];
  const float* Wo   = (const float*)d_in[3];
  const float* bo   = (const float*)d_in[4];
  float* out        = (float*)d_out;

  // workspace layout (bf16): Q | K | V^T | Y  (each B*H*L*Dh = 6,291,456 elems)
  const size_t QS = (size_t)B_ * H_ * L_ * DH_;
  bf16* ws  = (bf16*)d_ws;
  bf16* qws = ws;
  bf16* kws = ws + QS;
  bf16* vws = ws + 2 * QS;
  bf16* yws = ws + 3 * QS;

  // 1) fused QKV projection: M=8192, N=2304, K=768
  qkv_proj_wmma<<<dim3(2304 / 128, (B_ * L_) / 128), 256, 0, stream>>>(
      x, Wqkv, bqkv, qws, kws, vws);

  // 2) causal flash attention: one block per (64 query rows, b*H+h)
  flash_attn_wmma<<<dim3(L_ / 64, B_ * H_), 128, 0, stream>>>(
      qws, kws, vws, yws);

  // 3) output projection: M=8192, N=768, K=768
  out_proj_wmma<<<dim3(D_ / 128, (B_ * L_) / 128), 256, 0, stream>>>(
      yws, Wo, bo, out);
}
